// DecoderSaliency_AttModule_61727269978641
// MI455X (gfx1250) — matrix operand order repaired and checked by reading, not compiled, not run on hardware
//
#include <hip/hip_runtime.h>
#include <hip/hip_bf16.h>

// Problem dims
#define BB 128
#define RR 36
#define FF 2048
#define SS 4096
#define DD 1024
#define AAdim 512
#define EE 1024
#define VV 10000
#define LL 21
#define TT 20

typedef __bf16 bf16_t;
typedef __attribute__((ext_vector_type(16))) __bf16 v16bf;
typedef __attribute__((ext_vector_type(8)))  __bf16 v8bf;
typedef __attribute__((ext_vector_type(8)))  float  v8f;

__device__ __forceinline__ bf16_t f2bf(float f) {
  union { float f; unsigned u; } v; v.f = f;
  unsigned r = v.u + 0x7FFFu + ((v.u >> 16) & 1u);   // round-to-nearest-even
  unsigned short h = (unsigned short)(r >> 16);
  union { unsigned short s; bf16_t b; } o; o.s = h;
  return o.b;
}

__device__ __forceinline__ v16bf cat16(v8bf lo, v8bf hi) {
  return __builtin_shufflevector(lo, hi, 0, 1, 2, 3, 4, 5, 6, 7,
                                          8, 9, 10, 11, 12, 13, 14, 15);
}

// ---------------------------------------------------------------------------
// Skinny GEMM: C[M,N] (f32) = A[M,K](bf16) @ W[N,K](bf16)^T
//   + optional bias[N] + optional init0[M,N] + optional (+= C), optional
//   per-row masking (write 0.0 where t >= declen[m]).
// One wave computes a 16 x (16*NT) tile; NT is compile-time so the K-loop is
// branch-free and accumulators never get phi-copied. M % 16 == 0, K % 32 == 0.
// All fragment addresses derive directly from the A / W kernel arguments so
// address-space inference keeps every load a global_load_b128 (no flat).
// A-fragment: lane half h, lane&15 = row; K packing {k+8h.., k+16+8h..}.
// B-fragment: lane&15 = out column; K packing {k+8h.., k+16+8h..}.
// ---------------------------------------------------------------------------
template <int NT>
__global__ __launch_bounds__(32)
void gemm_bf16_wmma(const bf16_t* __restrict__ A, int lda,
                    const bf16_t* __restrict__ W, int ldw,
                    float* __restrict__ C, long long ldc,
                    const float* __restrict__ bias,
                    const float* __restrict__ init0, int ldi0,
                    int addC,
                    const int* __restrict__ declen, int t,
                    int nbase, int K)
{
  const int lane = threadIdx.x;
  const int half = lane >> 4;
  const int lm   = lane & 15;
  const int m0   = blockIdx.y << 4;
  const int n0   = nbase + (blockIdx.x << 6);

  v8f acc[NT];
#pragma unroll
  for (int j = 0; j < NT; ++j) {
    v8f z = {0.f, 0.f, 0.f, 0.f, 0.f, 0.f, 0.f, 0.f};
    acc[j] = z;
  }

  const bf16_t* ap    = A + (long long)(m0 + lm) * lda + half * 8;
  const bf16_t* wbase = W + (long long)(n0 + lm) * ldw + half * 8;
  const long long wstride = (long long)16 * ldw;   // row stride between subtiles

  for (int k = 0; k < K; k += 32) {
    v16bf a = cat16(*(const v8bf*)(ap + k), *(const v8bf*)(ap + k + 16));
#pragma unroll
    for (int j = 0; j < NT; ++j) {
      const bf16_t* wr = wbase + j * wstride + k;
      v16bf b = cat16(*(const v8bf*)(wr), *(const v8bf*)(wr + 16));
      acc[j] = __builtin_amdgcn_wmma_f32_16x16x32_bf16(
          false, a, false, b, (short)0, acc[j], false, false);
    }
  }

#pragma unroll
  for (int j = 0; j < NT; ++j) {
    int n = n0 + j * 16 + lm;
    float bv = bias ? bias[n] : 0.0f;
#pragma unroll
    for (int d = 0; d < 8; ++d) {
      int m = m0 + d + half * 8;
      float v = acc[j][d] + bv;
      if (init0) v += init0[(long long)m * ldi0 + n];
      long long ci = (long long)m * ldc + n;
      if (addC) v += C[ci];
      if (declen && t >= declen[m]) v = 0.0f;
      C[ci] = v;
    }
  }
}

// ---------------------------------------------------------------------------
// Stable descending argsort of sizes (rank-count), dec_len, gathered seq,
// plus the float-converted tail outputs (seq, dec_len, order).
// ---------------------------------------------------------------------------
__global__ void sort_order_kernel(const int* __restrict__ sizes,
                                  const int* __restrict__ seqs,
                                  int* __restrict__ order,
                                  int* __restrict__ declen,
                                  int* __restrict__ seq_ord,
                                  float* __restrict__ out_meta)
{
  int i = threadIdx.x;                 // 0..127
  int si = sizes[i];
  int rank = 0;
  for (int j = 0; j < BB; ++j) {
    int sj = sizes[j];
    if (sj > si || (sj == si && j < i)) rank++;
  }
  order[rank] = i;
  __syncthreads();
  int o  = order[i];
  int dl = sizes[o] - 1;
  declen[i] = dl;
  for (int l = 0; l < LL; ++l) {
    int v = seqs[o * LL + l];
    seq_ord[i * LL + l] = v;
    out_meta[i * LL + l] = (float)v;   // seq output (as float)
  }
  out_meta[BB * LL + i]      = (float)dl;  // dec_len output
  out_meta[BB * LL + BB + i] = (float)o;   // order output
}

__global__ void cvt_bf16(const float* __restrict__ in, bf16_t* __restrict__ out,
                         long long n)
{
  long long i = (long long)blockIdx.x * blockDim.x + threadIdx.x;
  long long stride = (long long)gridDim.x * blockDim.x;
  for (; i < n; i += stride) out[i] = f2bf(in[i]);
}

__global__ void gather_feats_kernel(const float* __restrict__ feats,
                                    const int* __restrict__ order,
                                    bf16_t* __restrict__ out)
{
  long long idx = (long long)blockIdx.x * 256 + threadIdx.x;
  if (idx >= (long long)BB * RR * FF) return;
  long long b = idx / ((long long)RR * FF);
  long long rem = idx - b * RR * FF;
  out[idx] = f2bf(feats[(long long)order[b] * RR * FF + rem]);
}

__global__ void gather_sal_kernel(const float* __restrict__ sal,
                                  const int* __restrict__ order,
                                  bf16_t* __restrict__ out)
{
  int idx = blockIdx.x * 256 + threadIdx.x;
  if (idx >= BB * SS) return;
  int b = idx >> 12, f = idx & (SS - 1);
  out[idx] = f2bf(sal[(long long)order[b] * SS + f]);
}

__global__ void feats_avg_kernel(const float* __restrict__ feats,
                                 const int* __restrict__ order,
                                 float* __restrict__ favg,
                                 bf16_t* __restrict__ favg_bf)
{
  int idx = blockIdx.x * 256 + threadIdx.x;
  if (idx >= BB * FF) return;
  int b = idx >> 11, f = idx & (FF - 1);
  const float* fb = feats + (long long)order[b] * RR * FF + f;
  float s = 0.0f;
#pragma unroll 4
  for (int r = 0; r < RR; ++r) s += fb[(long long)r * FF];
  s *= (1.0f / (float)RR);
  favg[idx] = s;
  favg_bf[idx] = f2bf(s);
}

__global__ void gather_emb_kernel(const float* __restrict__ emb,
                                  const int* __restrict__ seq_ord,
                                  int t, bf16_t* __restrict__ out)
{
  int idx = blockIdx.x * 256 + threadIdx.x;
  if (idx >= BB * EE) return;
  int b = idx >> 10, e = idx & (EE - 1);
  int tok = seq_ord[b * LL + t];
  out[idx] = f2bf(emb[(long long)tok * EE + e]);
}

__global__ void zero_states_kernel(float* h1, float* c1, float* h2, float* c2,
                                   bf16_t* h1b, bf16_t* h2b)
{
  int i = blockIdx.x * 256 + threadIdx.x;
  if (i >= BB * DD) return;
  h1[i] = 0.f; c1[i] = 0.f; h2[i] = 0.f; c2[i] = 0.f;
  h1b[i] = f2bf(0.f); h2b[i] = f2bf(0.f);
}

// Pointwise LSTM cell (torch gate order i,f,g,o). States frozen where masked.
__global__ void lstm_cell_kernel(const float* __restrict__ g,
                                 float* __restrict__ h, float* __restrict__ c,
                                 bf16_t* __restrict__ hbf,
                                 const int* __restrict__ declen, int t)
{
  int idx = blockIdx.x * 256 + threadIdx.x;
  if (idx >= BB * DD) return;
  int b = idx >> 10, d = idx & (DD - 1);
  const float* gr = g + (long long)b * 4 * DD;
  float hv = h[idx], cv = c[idx];
  if (t < declen[b]) {
    float gi = gr[d], gf = gr[DD + d], gg = gr[2 * DD + d], go = gr[3 * DD + d];
    float si = 1.f / (1.f + __expf(-gi));
    float sf = 1.f / (1.f + __expf(-gf));
    float so = 1.f / (1.f + __expf(-go));
    cv = sf * cv + si * tanhf(gg);
    hv = so * tanhf(cv);
    c[idx] = cv; h[idx] = hv;
  }
  hbf[idx] = f2bf(hv);
}

// Attention: one block per batch row. 37 logits -> softmax -> weighted sum.
__global__ __launch_bounds__(256)
void attention_kernel(const float* __restrict__ h1att,
                      const float* __restrict__ img_att,
                      const float* __restrict__ sal_ab,
                      const float* __restrict__ Wa, const float* __restrict__ ba,
                      const float* __restrict__ feats, const int* __restrict__ order,
                      const float* __restrict__ sal_w,
                      float* __restrict__ aw, bf16_t* __restrict__ awbf)
{
  int b = blockIdx.x;
  __shared__ float sh[AAdim];
  __shared__ float slog[RR + 1];
  __shared__ float sw[RR + 1];
  int tid = threadIdx.x;
  for (int i = tid; i < AAdim; i += 256) sh[i] = h1att[(long long)b * AAdim + i];
  __syncthreads();

  int wave = tid >> 5, lane = tid & 31;
  for (int item = wave; item < RR + 1; item += 8) {
    const float* basep = (item < RR)
        ? (img_att + ((long long)b * RR + item) * AAdim)
        : (sal_ab + (long long)b * AAdim);
    float s = 0.f;
    for (int j = lane; j < AAdim; j += 32) {
      float v = sh[j] + basep[j];
      v = v > 0.f ? v : 0.f;
      s += v * Wa[j];
    }
#pragma unroll
    for (int off = 16; off > 0; off >>= 1) s += __shfl_xor(s, off, 32);
    if (lane == 0) slog[item] = s + ba[0];
  }
  __syncthreads();
  if (tid == 0) {
    float mx = slog[0];
    for (int i = 1; i < RR + 1; ++i) mx = fmaxf(mx, slog[i]);
    float sum = 0.f;
    for (int i = 0; i < RR + 1; ++i) { float e = __expf(slog[i] - mx); sw[i] = e; sum += e; }
    float inv = 1.f / sum;
    for (int i = 0; i < RR + 1; ++i) sw[i] *= inv;
  }
  __syncthreads();

  const float* fb = feats + (long long)order[b] * RR * FF;
  float wsal = sw[RR];
  for (int f = tid; f < FF; f += 256) {
    float s = sal_w[(long long)b * FF + f] * wsal;
#pragma unroll 4
    for (int r = 0; r < RR; ++r) s += fb[(long long)r * FF + f] * sw[r];
    aw[(long long)b * FF + f] = s;
    awbf[(long long)b * FF + f] = f2bf(s);
  }
}

// ---------------------------------------------------------------------------
extern "C" void kernel_launch(void* const* d_in, const int* in_sizes, int n_in,
                              void* d_out, int out_size, void* d_ws, size_t ws_size,
                              hipStream_t stream)
{
  (void)in_sizes; (void)n_in; (void)out_size; (void)ws_size;

  const float* feats    = (const float*)d_in[0];
  const float* sal      = (const float*)d_in[1];
  const int*   seqs     = (const int*)  d_in[2];
  const int*   sizes    = (const int*)  d_in[3];
  const float* emb      = (const float*)d_in[4];
  const float* td_wih   = (const float*)d_in[5];
  const float* td_whh   = (const float*)d_in[6];
  const float* td_b     = (const float*)d_in[7];
  const float* lang_wih = (const float*)d_in[8];
  const float* lang_whh = (const float*)d_in[9];
  const float* lang_b   = (const float*)d_in[10];
  const float* Waf      = (const float*)d_in[11];
  const float* baf      = (const float*)d_in[12];
  const float* Wad      = (const float*)d_in[13];
  const float* bad      = (const float*)d_in[14];
  const float* Wsal     = (const float*)d_in[15];
  const float* bsal     = (const float*)d_in[16];
  const float* Was      = (const float*)d_in[17];
  const float* bas      = (const float*)d_in[18];
  const float* Wa       = (const float*)d_in[19];
  const float* ba       = (const float*)d_in[20];
  const float* Wout     = (const float*)d_in[21];
  const float* bout     = (const float*)d_in[22];

  // ---- carve workspace (256B aligned chunks) ----
  char* p = (char*)d_ws;
  auto alloc = [&](size_t bytes) -> char* {
    char* r = p; p += (bytes + 255) & ~(size_t)255; return r;
  };
  int*    order       = (int*)   alloc(BB * sizeof(int));
  int*    declen      = (int*)   alloc(BB * sizeof(int));
  int*    seq_ord     = (int*)   alloc((size_t)BB * LL * sizeof(int));
  bf16_t* td_wih_bf   = (bf16_t*)alloc((size_t)4 * DD * (EE + FF + DD) * 2);
  bf16_t* td_whh_bf   = (bf16_t*)alloc((size_t)4 * DD * DD * 2);
  bf16_t* lang_wih_bf = (bf16_t*)alloc((size_t)4 * DD * (FF + DD) * 2);
  bf16_t* lang_whh_bf = (bf16_t*)alloc((size_t)4 * DD * DD * 2);
  bf16_t* Wout_bf     = (bf16_t*)alloc((size_t)VV * DD * 2);
  bf16_t* Waf_bf      = (bf16_t*)alloc((size_t)AAdim * FF * 2);
  bf16_t* Wad_bf      = (bf16_t*)alloc((size_t)AAdim * DD * 2);
  bf16_t* Wsal_bf     = (bf16_t*)alloc((size_t)FF * SS * 2);
  bf16_t* Was_bf      = (bf16_t*)alloc((size_t)AAdim * FF * 2);
  bf16_t* feats_bf    = (bf16_t*)alloc((size_t)BB * RR * FF * 2);
  bf16_t* sal_bf      = (bf16_t*)alloc((size_t)BB * SS * 2);
  bf16_t* embt_bf     = (bf16_t*)alloc((size_t)BB * EE * 2);
  float*  favg        = (float*) alloc((size_t)BB * FF * 4);
  bf16_t* favg_bf     = (bf16_t*)alloc((size_t)BB * FF * 2);
  float*  g1_base     = (float*) alloc((size_t)BB * 4 * DD * 4);
  float*  img_att     = (float*) alloc((size_t)BB * RR * AAdim * 4);
  float*  sal_w       = (float*) alloc((size_t)BB * FF * 4);
  bf16_t* sal_w_bf    = (bf16_t*)alloc((size_t)BB * FF * 2);
  float*  sal_ab      = (float*) alloc((size_t)BB * AAdim * 4);
  float*  h1          = (float*) alloc((size_t)BB * DD * 4);
  float*  c1          = (float*) alloc((size_t)BB * DD * 4);
  float*  h2          = (float*) alloc((size_t)BB * DD * 4);
  float*  c2          = (float*) alloc((size_t)BB * DD * 4);
  bf16_t* h1_bf       = (bf16_t*)alloc((size_t)BB * DD * 2);
  bf16_t* h2_bf       = (bf16_t*)alloc((size_t)BB * DD * 2);
  float*  g1          = (float*) alloc((size_t)BB * 4 * DD * 4);
  float*  g2          = (float*) alloc((size_t)BB * 4 * DD * 4);
  float*  h1att       = (float*) alloc((size_t)BB * AAdim * 4);
  float*  aw          = (float*) alloc((size_t)BB * FF * 4);
  bf16_t* aw_bf       = (bf16_t*)alloc((size_t)BB * FF * 2);

  float* preds    = (float*)d_out;                      // [B, T, V]
  float* out_meta = preds + (size_t)BB * TT * VV;       // seq, dec_len, order

  auto cvt = [&](const float* in, bf16_t* out, long long n) {
    int blocks = (int)((n + 2047) / 2048);
    if (blocks > 4096) blocks = 4096;
    cvt_bf16<<<blocks, 256, 0, stream>>>(in, out, n);
  };
  // Dispatch: full 64-col blocks with NT=4, plus one tail kernel for N%64.
  auto gemm = [&](const bf16_t* Ap, int lda, const bf16_t* Wp, int ldw,
                  float* Cp, long long ldc, const float* biasp,
                  const float* init0p, int ldi0, int addC,
                  const int* dl, int t, int M, int N, int K) {
    int full = N >> 6;
    if (full)
      gemm_bf16_wmma<4><<<dim3(full, M / 16), dim3(32), 0, stream>>>(
          Ap, lda, Wp, ldw, Cp, ldc, biasp, init0p, ldi0, addC, dl, t, 0, K);
    int rem = (N & 63) >> 4;
    int nbase = full << 6;
    if (rem == 1)
      gemm_bf16_wmma<1><<<dim3(1, M / 16), dim3(32), 0, stream>>>(
          Ap, lda, Wp, ldw, Cp, ldc, biasp, init0p, ldi0, addC, dl, t, nbase, K);
    else if (rem == 2)
      gemm_bf16_wmma<2><<<dim3(1, M / 16), dim3(32), 0, stream>>>(
          Ap, lda, Wp, ldw, Cp, ldc, biasp, init0p, ldi0, addC, dl, t, nbase, K);
    else if (rem == 3)
      gemm_bf16_wmma<3><<<dim3(1, M / 16), dim3(32), 0, stream>>>(
          Ap, lda, Wp, ldw, Cp, ldc, biasp, init0p, ldi0, addC, dl, t, nbase, K);
  };

  // ---- preprocessing ----
  sort_order_kernel<<<1, BB, 0, stream>>>(sizes, seqs, order, declen, seq_ord, out_meta);
  cvt(td_wih,   td_wih_bf,   (long long)4 * DD * (EE + FF + DD));
  cvt(td_whh,   td_whh_bf,   (long long)4 * DD * DD);
  cvt(lang_wih, lang_wih_bf, (long long)4 * DD * (FF + DD));
  cvt(lang_whh, lang_whh_bf, (long long)4 * DD * DD);
  cvt(Wout,     Wout_bf,     (long long)VV * DD);
  cvt(Waf,      Waf_bf,      (long long)AAdim * FF);
  cvt(Wad,      Wad_bf,      (long long)AAdim * DD);
  cvt(Wsal,    Wsal_bf,      (long long)FF * SS);
  cvt(Was,      Was_bf,      (long long)AAdim * FF);

  gather_feats_kernel<<<(int)(((long long)BB * RR * FF + 255) / 256), 256, 0, stream>>>(feats, order, feats_bf);
  gather_sal_kernel<<<(BB * SS + 255) / 256, 256, 0, stream>>>(sal, order, sal_bf);
  feats_avg_kernel<<<(BB * FF + 255) / 256, 256, 0, stream>>>(feats, order, favg, favg_bf);
  zero_states_kernel<<<(BB * DD + 255) / 256, 256, 0, stream>>>(h1, c1, h2, c2, h1_bf, h2_bf);

  // img_att = feats_ord @ Waf^T + baf           [B*R, A]
  gemm(feats_bf, FF, Waf_bf, FF, img_att, AAdim, baf, nullptr, 0, 0, nullptr, 0, BB * RR, AAdim, FF);
  // sal_w = sal_ord @ Wsal^T + bsal             [B, F]
  gemm(sal_bf, SS, Wsal_bf, SS, sal_w, FF, bsal, nullptr, 0, 0, nullptr, 0, BB, FF, SS);
  cvt(sal_w, sal_w_bf, (long long)BB * FF);
  // sal_att_base = sal_w @ Was^T + bas          [B, A]
  gemm(sal_w_bf, FF, Was_bf, FF, sal_ab, AAdim, bas, nullptr, 0, 0, nullptr, 0, BB, AAdim, FF);
  // g1_base = featsAvg @ td_wih[:, D:D+F]^T + td_b   (time-invariant gate part)
  gemm(favg_bf, FF, td_wih_bf + DD, EE + FF + DD, g1_base, 4 * DD, td_b,
       nullptr, 0, 0, nullptr, 0, BB, 4 * DD, FF);

  // ---- recurrence ----
  for (int t = 0; t < TT; ++t) {
    gather_emb_kernel<<<(BB * EE + 255) / 256, 256, 0, stream>>>(emb, seq_ord, t, embt_bf);
    // g1 = h2 @ Wih[:, :D]^T + g1_base
    gemm(h2_bf, DD, td_wih_bf, EE + FF + DD, g1, 4 * DD, nullptr,
         g1_base, 4 * DD, 0, nullptr, 0, BB, 4 * DD, DD);
    // g1 += emb_t @ Wih[:, D+F:]^T
    gemm(embt_bf, EE, td_wih_bf + (DD + FF), EE + FF + DD, g1, 4 * DD, nullptr,
         nullptr, 0, 1, nullptr, 0, BB, 4 * DD, EE);
    // g1 += h1 @ td_whh^T
    gemm(h1_bf, DD, td_whh_bf, DD, g1, 4 * DD, nullptr,
         nullptr, 0, 1, nullptr, 0, BB, 4 * DD, DD);
    lstm_cell_kernel<<<(BB * DD + 255) / 256, 256, 0, stream>>>(g1, h1, c1, h1_bf, declen, t);
    // h1_att = h1 @ Wad^T + bad
    gemm(h1_bf, DD, Wad_bf, DD, h1att, AAdim, bad, nullptr, 0, 0, nullptr, 0, BB, AAdim, DD);
    attention_kernel<<<BB, 256, 0, stream>>>(h1att, img_att, sal_ab, Wa, ba,
                                             feats, order, sal_w, aw, aw_bf);
    // g2 = aw @ lang_wih[:, :F]^T + lang_b
    gemm(aw_bf, FF, lang_wih_bf, FF + DD, g2, 4 * DD, lang_b,
         nullptr, 0, 0, nullptr, 0, BB, 4 * DD, FF);
    // g2 += h1 @ lang_wih[:, F:]^T
    gemm(h1_bf, DD, lang_wih_bf + FF, FF + DD, g2, 4 * DD, nullptr,
         nullptr, 0, 1, nullptr, 0, BB, 4 * DD, DD);
    // g2 += h2 @ lang_whh^T      (old h2)
    gemm(h2_bf, DD, lang_whh_bf, DD, g2, 4 * DD, nullptr,
         nullptr, 0, 1, nullptr, 0, BB, 4 * DD, DD);
    lstm_cell_kernel<<<(BB * DD + 255) / 256, 256, 0, stream>>>(g2, h2, c2, h2_bf, declen, t);
    // preds[:, t, :] = mask ? h2 @ Wout^T + bout : 0
    gemm(h2_bf, DD, Wout_bf, DD, preds + (long long)t * VV, (long long)TT * VV,
         bout, nullptr, 0, 0, declen, t, BB, VV, DD);
  }
}
